// ConvModel_58720792871499
// MI455X (gfx1250) — matrix operand order repaired and checked by reading, not compiled, not run on hardware
//
#include <hip/hip_runtime.h>
#include <hip/hip_bf16.h>

// CDNA5 (gfx1250) wave32 WMMA implicit-GEMM conv2d 3x3 s1 p1.
// GEMM view: M = C_OUT = 128, N = pixels = 16*224*224, K = C_IN*9 = 576.

typedef __attribute__((ext_vector_type(16))) __bf16 v16bf;
typedef __attribute__((ext_vector_type(8)))  float  v8f;

#define C_IN   64
#define C_OUT  128
#define HW     224
#define TILE_W 32          // output pixels per block (224 = 7 * 32)
#define KCHUNK 18          // 576 / 32 K-chunks
#define LDS_CI 72          // padded ci stride in LDS (bank spread, 144B = 16B-aligned)

union V16 { uint4 q[2]; v16bf v; };

// ---------------------------------------------------------------------------
// Kernel 1: repack fp32 weights [co][ci][3][3] -> bf16 A-fragment layout.
// packed[coTile(8)][chunk(18)][lane(32)][e(16)], chunk c: tap r = c>>1,
// ci = (c&1)*32 + K,  K = e + 8*((e>>3) + (lane>>4))   (ISA 7.12.2, 16-bit A 16x32)
// ---------------------------------------------------------------------------
__global__ __launch_bounds__(256) void weight_repack_kernel(
    const float* __restrict__ w, __bf16* __restrict__ packed) {
  int p = blockIdx.x * 256 + threadIdx.x;        // 8*18*32*16 = 73728 elements
  if (p >= 8 * KCHUNK * 32 * 16) return;
  int e    = p & 15;
  int lane = (p >> 4) & 31;
  int c    = (p >> 9) % KCHUNK;
  int t    = p / (KCHUNK * 512);                 // co tile
  int co   = t * 16 + (lane & 15);
  int K    = e + 8 * ((e >> 3) + (lane >> 4));
  int r    = c >> 1;                             // tap: kh*3+kw
  int ci   = (c & 1) * 32 + K;
  // weight flat index: ((co*64 + ci)*3 + kh)*3 + kw = co*576 + ci*9 + r
  packed[p] = (__bf16)w[(size_t)co * 576 + ci * 9 + r];
}

// ---------------------------------------------------------------------------
// Kernel 2: each block -> 128 co x 32 pixels of one (n, h) output row.
// 8 waves: wave i owns co tile i (16 co) x 32 px (two 16x16 WMMA tiles).
// ---------------------------------------------------------------------------
__global__ __launch_bounds__(256) void conv_wmma_kernel(
    const float* __restrict__ x, const __bf16* __restrict__ wpack,
    const float* __restrict__ bias, float* __restrict__ out) {
  __shared__ __bf16 xs[3 * 34 * LDS_CI];         // [kh][w(34)][ci(64, pad 72)] bf16

  int b   = blockIdx.x;
  int seg = b % 7;
  int h   = (b / 7) % HW;
  int n   = b / (7 * HW);
  int w0  = seg * TILE_W;
  int tid = threadIdx.x;

  // ---- stage input halo (64ci x 3rows x 34w) into LDS, fp32 -> bf16 ----
  for (int i = tid; i < 3 * C_IN * 34; i += 256) {
    int wi  = i % 34;
    int t   = i / 34;
    int ci  = t & 63;
    int kh  = t >> 6;
    int row = h - 1 + kh;
    int gw  = w0 - 1 + wi;
    float v = 0.0f;
    if ((unsigned)row < (unsigned)HW && (unsigned)gw < (unsigned)HW)
      v = x[(((size_t)n * C_IN + ci) * HW + row) * HW + gw];
    xs[(kh * 34 + wi) * LDS_CI + ci] = (__bf16)v;
  }
  __syncthreads();

  // wave32: EXEC is all ones here (uniform control flow) as WMMA requires.
  int wave = tid >> 5;
  int lane = tid & 31;
  int px   = lane & 15;       // B/D column (pixel within 16-wide tile)
  int hi   = lane >> 4;       // upper half-wave selector

  v8f acc0 = {};              // pixels w0+0  .. w0+15
  v8f acc1 = {};              // pixels w0+16 .. w0+31

  const __bf16* wbase = wpack + (size_t)wave * KCHUNK * 512 + lane * 16;

#pragma unroll
  for (int r = 0; r < 9; ++r) {
    const int kh = r / 3, kw = r % 3;
#pragma unroll
    for (int half = 0; half < 2; ++half) {
      const int c = r * 2 + half;
      // A fragment: coalesced 32B/lane from packed weights (L2-resident)
      v16bf A = *(const v16bf*)(wbase + c * 512);

      // B fragments from LDS: lane holds two runs of 8 ci values
      // (lanes 0-15: K=0..7,16..23, lanes 16-31: K=8..15,24..31), two ds_load_b128 each.
      const int base = half * 32 + hi * 8;
      int offB0 = (kh * 34 + (px + kw)) * LDS_CI + base;
      int offB1 = (kh * 34 + (px + 16 + kw)) * LDS_CI + base;

      V16 b0, b1;
      b0.q[0] = *(const uint4*)&xs[offB0];
      b0.q[1] = *(const uint4*)&xs[offB0 + 16];
      b1.q[0] = *(const uint4*)&xs[offB1];
      b1.q[1] = *(const uint4*)&xs[offB1 + 16];

      acc0 = __builtin_amdgcn_wmma_f32_16x16x32_bf16(
          false, A, false, b0.v, (short)0, acc0, false, false);
      acc1 = __builtin_amdgcn_wmma_f32_16x16x32_bf16(
          false, A, false, b1.v, (short)0, acc1, false, false);
    }
  }

  // ---- epilogue: D layout VGPR r -> M = r + 8*hi, lane -> N = px ----
#pragma unroll
  for (int rr = 0; rr < 8; ++rr) {
    int co = wave * 16 + rr + hi * 8;
    float bv = bias[co];
    size_t o = (((size_t)n * C_OUT + co) * HW + h) * HW + w0;
    out[o + px]      = acc0[rr] + bv;
    out[o + 16 + px] = acc1[rr] + bv;
  }
}

// ---------------------------------------------------------------------------
extern "C" void kernel_launch(void* const* d_in, const int* in_sizes, int n_in,
                              void* d_out, int out_size, void* d_ws, size_t ws_size,
                              hipStream_t stream) {
  const float* x    = (const float*)d_in[0];   // [16,64,224,224]
  const float* w    = (const float*)d_in[1];   // [128,64,3,3]
  const float* bias = (const float*)d_in[2];   // [128]
  float* out        = (float*)d_out;           // [16,128,224,224]
  __bf16* wpack     = (__bf16*)d_ws;           // 147456 bytes of scratch

  weight_repack_kernel<<<(8 * KCHUNK * 32 * 16 + 255) / 256, 256, 0, stream>>>(w, wpack);

  const int nblocks = 16 * HW * (HW / TILE_W); // 25088
  conv_wmma_kernel<<<nblocks, 256, 0, stream>>>(x, wpack, bias, out);
}
